// GRU_SRU_64871186039268
// MI455X (gfx1250) — compile-verified, matrix operand order checked by reading, not run on hardware
//
#include <hip/hip_runtime.h>

#define T_N 512
#define B_N 64
#define H_N 512
#define L_N 2
#define H2_N 1024

typedef __attribute__((ext_vector_type(16))) _Float16 v16h;
typedef __attribute__((ext_vector_type(8)))  float    v8f;

__device__ __forceinline__ float sigmoidf_(float v) {
  return 1.0f / (1.0f + __expf(-v));
}

// ---------------------------------------------------------------------------
// WMMA 16x16x32 f16 fragment loader (wave32, layouts per CDNA5 ISA 7.12.2).
// Works for both operands of  in @ W^T :
//   A fragment: row = m (batch row), lane holds row m_base + (lane&15)
//   B fragment: "row" = weight row n == output column n
// halves 0..7  hold K = k0      + (lane&16 ? 8 : 0) + j
// halves 8..15 hold K = k0 + 16 + (lane&16 ? 8 : 0) + j
// C/D: vgpr v, lane l -> M = v + 8*(l>=16), N = l&15.
// Source matrices are row-major f16 with leading dimension H_N.
// ---------------------------------------------------------------------------
__device__ __forceinline__ v16h frag_f16(const _Float16* __restrict__ M,
                                         int row, int k0, int lane) {
  const _Float16* p = M + (size_t)row * H_N + k0 + ((lane & 16) ? 8 : 0);
  v16h f;
#pragma unroll
  for (int j = 0; j < 8; ++j) {
    f[j]     = p[j];
    f[j + 8] = p[j + 16];
  }
  return f;
}

// One wave computes a full 64x16 strip (4 m-tiles) for output column strip
// `col`: B fragment loaded once per K chunk, reused for 4 WMMAs.
__device__ __forceinline__ void gemm_strip(const _Float16* __restrict__ A,  // [B_N][H_N]
                                           const _Float16* __restrict__ W,  // [N][H_N]
                                           int col, int lane, v8f acc[4]) {
#pragma unroll 2
  for (int k = 0; k < H_N; k += 32) {
    const v16h b = frag_f16(W, col, k, lane);
#pragma unroll
    for (int mt = 0; mt < 4; ++mt) {
      const v16h a = frag_f16(A, mt * 16 + (lane & 15), k, lane);
      acc[mt] = __builtin_amdgcn_wmma_f32_16x16x32_f16(false, a, false, b,
                                                       (short)0, acc[mt],
                                                       false, false);
    }
  }
}

// ---------------------------------------------------------------------------
// Prep kernels: f32 -> f16 conversion; weights split so K is contiguous.
// ---------------------------------------------------------------------------
__global__ void prep_wall_k(const float* __restrict__ W_all,
                            _Float16* __restrict__ Wx, _Float16* __restrict__ Wh) {
  int idx = blockIdx.x * blockDim.x + threadIdx.x;     // over L*2H*H
  if (idx >= L_N * H2_N * H_N) return;
  int k = idx % H_N;
  int n = (idx / H_N) % H2_N;
  int l = idx / (H_N * H2_N);
  const float* row = W_all + ((size_t)l * H2_N + n) * H2_N;
  Wx[idx] = (_Float16)row[k];
  Wh[idx] = (_Float16)row[H_N + k];
}

__global__ void prep_wn_k(const float* __restrict__ W_n,
                          _Float16* __restrict__ Wnx, _Float16* __restrict__ Wnh) {
  int idx = blockIdx.x * blockDim.x + threadIdx.x;     // over L*H*H
  if (idx >= L_N * H_N * H_N) return;
  int k = idx % H_N;
  int n = (idx / H_N) % H_N;
  int l = idx / (H_N * H_N);
  const float* row = W_n + ((size_t)l * H_N + n) * H2_N;
  Wnx[idx] = (_Float16)row[k];
  Wnh[idx] = (_Float16)row[H_N + k];
}

__global__ void prep_wtx_k(const float* __restrict__ W_tx,
                           _Float16* __restrict__ Wtx) {
  int idx = blockIdx.x * blockDim.x + threadIdx.x;     // over L*H*H
  if (idx >= L_N * H_N * H_N) return;
  Wtx[idx] = (_Float16)W_tx[idx];
}

__global__ void prep_x16_k(const float* __restrict__ x, _Float16* __restrict__ x16) {
  int idx = blockIdx.x * blockDim.x + threadIdx.x;     // over T*B*H
  if (idx >= T_N * B_N * H_N) return;
  x16[idx] = (_Float16)x[idx];
}

// ---------------------------------------------------------------------------
// Step kernel 1: fused projections for one layer at one time step.
//   n strips [0,64)    : gates = x@Wx^T + h@Wh^T + b_all
//                        cols < H  -> store z = sigmoid(g)
//                        cols >= H -> store rh16 = f16(sigmoid(g) * h)
//   n strips [64,96)   : txb = x@Wtx^T + b_tx              (f32)
//   n strips [96,128)  : nxb = x@Wnx^T + b_n               (f32)
// 128 waves (one per 64x16 strip) = 16 blocks x 256 threads.
// ---------------------------------------------------------------------------
__global__ __launch_bounds__(256) void step_s1(
    const _Float16* __restrict__ x16, const _Float16* __restrict__ h16,
    const float* __restrict__ hf,
    const _Float16* __restrict__ Wx, const _Float16* __restrict__ Wh,
    const _Float16* __restrict__ Wtx, const _Float16* __restrict__ Wnx,
    const float* __restrict__ b_all, const float* __restrict__ b_tx,
    const float* __restrict__ b_n,
    float* __restrict__ zbuf, _Float16* __restrict__ rh16,
    float* __restrict__ txb, float* __restrict__ nxb) {
  const int lane = threadIdx.x & 31;
  const int wave = threadIdx.x >> 5;
  const int strip = blockIdx.x * 8 + wave;  // 0..127
  const int n_base = strip * 16;
  const int colL = lane & 15;
  const int mhi = (lane >> 4) << 3;         // +8 for lanes 16..31

  v8f acc[4];
  if (n_base < H2_N) {                      // ---- gates section ----
    const int col = n_base + colL;
    const float bias = b_all[col];
#pragma unroll
    for (int mt = 0; mt < 4; ++mt)
#pragma unroll
      for (int v = 0; v < 8; ++v) acc[mt][v] = bias;
    gemm_strip(x16, Wx, col, lane, acc);
    gemm_strip(h16, Wh, col, lane, acc);
    if (n_base < H_N) {                     // z half
#pragma unroll
      for (int mt = 0; mt < 4; ++mt) {
        const int mrow = mt * 16 + mhi;
#pragma unroll
        for (int v = 0; v < 8; ++v)
          zbuf[(size_t)(mrow + v) * H_N + col] = sigmoidf_(acc[mt][v]);
      }
    } else {                                // r half -> rh16
      const int c = col - H_N;
#pragma unroll
      for (int mt = 0; mt < 4; ++mt) {
        const int mrow = mt * 16 + mhi;
#pragma unroll
        for (int v = 0; v < 8; ++v) {
          const size_t o = (size_t)(mrow + v) * H_N + c;
          rh16[o] = (_Float16)(sigmoidf_(acc[mt][v]) * hf[o]);
        }
      }
    }
  } else if (n_base < H2_N + H_N) {         // ---- tx section ----
    const int c = n_base - H2_N + colL;
    const float bias = b_tx[c];
#pragma unroll
    for (int mt = 0; mt < 4; ++mt)
#pragma unroll
      for (int v = 0; v < 8; ++v) acc[mt][v] = bias;
    gemm_strip(x16, Wtx, c, lane, acc);
#pragma unroll
    for (int mt = 0; mt < 4; ++mt) {
      const int mrow = mt * 16 + mhi;
#pragma unroll
      for (int v = 0; v < 8; ++v)
        txb[(size_t)(mrow + v) * H_N + c] = acc[mt][v];
    }
  } else {                                  // ---- nx section ----
    const int c = n_base - H2_N - H_N + colL;
    const float bias = b_n[c];
#pragma unroll
    for (int mt = 0; mt < 4; ++mt)
#pragma unroll
      for (int v = 0; v < 8; ++v) acc[mt][v] = bias;
    gemm_strip(x16, Wnx, c, lane, acc);
#pragma unroll
    for (int mt = 0; mt < 4; ++mt) {
      const int mrow = mt * 16 + mhi;
#pragma unroll
      for (int v = 0; v < 8; ++v)
        nxb[(size_t)(mrow + v) * H_N + c] = acc[mt][v];
    }
  }
}

// ---------------------------------------------------------------------------
// Step kernel 2: nh = rh16 @ Wnh^T (pure f16 GEMM), then the gate epilogue:
//   n  = tanh(tx * (nx + nh));  h' = (1-z)*n + z*h
// writes h' as f32 and f16 (and optionally the sequence output).
// 32 waves (one per 64x16 strip) = 4 blocks x 256 threads.
// ---------------------------------------------------------------------------
__global__ __launch_bounds__(256) void step_s2(
    const _Float16* __restrict__ rh16, const float* __restrict__ h,
    const _Float16* __restrict__ Wnh,
    const float* __restrict__ zbuf, const float* __restrict__ txb,
    const float* __restrict__ nxb,
    float* __restrict__ h_out, _Float16* __restrict__ h_out16,
    float* __restrict__ seq_out) {
  const int lane = threadIdx.x & 31;
  const int wave = threadIdx.x >> 5;
  const int strip = blockIdx.x * 8 + wave;  // 0..31
  const int col = strip * 16 + (lane & 15);
  const int mhi = (lane >> 4) << 3;

  v8f acc[4] = {};
  gemm_strip(rh16, Wnh, col, lane, acc);

#pragma unroll
  for (int mt = 0; mt < 4; ++mt) {
    const int mrow = mt * 16 + mhi;
#pragma unroll
    for (int v = 0; v < 8; ++v) {
      const size_t o = (size_t)(mrow + v) * H_N + col;
      const float z    = zbuf[o];
      const float npre = nxb[o] + acc[mt][v];
      const float nv   = tanhf(txb[o] * npre);
      const float hn   = (1.0f - z) * nv + z * h[o];
      h_out[o]   = hn;
      h_out16[o] = (_Float16)hn;
      if (seq_out) seq_out[o] = hn;
    }
  }
}

// ---------------------------------------------------------------------------
extern "C" void kernel_launch(void* const* d_in, const int* in_sizes, int n_in,
                              void* d_out, int out_size, void* d_ws, size_t ws_size,
                              hipStream_t stream) {
  (void)in_sizes; (void)n_in; (void)out_size; (void)ws_size;

  const float* x     = (const float*)d_in[0];   // [T,B,H]
  const float* W_all = (const float*)d_in[1];   // [L,2H,2H]
  const float* b_all = (const float*)d_in[2];   // [L,2H]
  const float* W_n   = (const float*)d_in[3];   // [L,H,2H]
  const float* b_n   = (const float*)d_in[4];   // [L,H]
  const float* W_tx  = (const float*)d_in[5];   // [L,H,H]
  const float* b_tx  = (const float*)d_in[6];   // [L,H]
  float* out = (float*)d_out;                   // [T,B,H] ++ [L,B,H] ++ [L,B,H]

  // ---- workspace bump allocation (256B aligned) ----
  char* p = (char*)d_ws;
  auto alloc = [&](size_t bytes) -> void* {
    void* r = (void*)p;
    p += (bytes + 255) & ~(size_t)255;
    return r;
  };
  _Float16* Wx    = (_Float16*)alloc((size_t)L_N * H2_N * H_N * sizeof(_Float16));
  _Float16* Wh    = (_Float16*)alloc((size_t)L_N * H2_N * H_N * sizeof(_Float16));
  _Float16* Wnx   = (_Float16*)alloc((size_t)L_N * H_N * H_N * sizeof(_Float16));
  _Float16* Wnh   = (_Float16*)alloc((size_t)L_N * H_N * H_N * sizeof(_Float16));
  _Float16* Wtx   = (_Float16*)alloc((size_t)L_N * H_N * H_N * sizeof(_Float16));
  _Float16* x16   = (_Float16*)alloc((size_t)T_N * B_N * H_N * sizeof(_Float16));
  float*    zbuf  = (float*)alloc((size_t)B_N * H_N * sizeof(float));
  _Float16* rh16  = (_Float16*)alloc((size_t)B_N * H_N * sizeof(_Float16));
  float*    txb   = (float*)alloc((size_t)B_N * H_N * sizeof(float));
  float*    nxb   = (float*)alloc((size_t)B_N * H_N * sizeof(float));
  float*    hbuf  = (float*)alloc((size_t)2 * L_N * B_N * H_N * sizeof(float));
  _Float16* hbuf16= (_Float16*)alloc((size_t)2 * L_N * B_N * H_N * sizeof(_Float16));

  // ---- prep (re-run every call; deterministic) ----
  {
    int tot = L_N * H2_N * H_N;
    prep_wall_k<<<(tot + 255) / 256, 256, 0, stream>>>(W_all, Wx, Wh);
    tot = L_N * H_N * H_N;
    prep_wn_k<<<(tot + 255) / 256, 256, 0, stream>>>(W_n, Wnx, Wnh);
    prep_wtx_k<<<(tot + 255) / 256, 256, 0, stream>>>(W_tx, Wtx);
    tot = T_N * B_N * H_N;
    prep_x16_k<<<(tot + 255) / 256, 256, 0, stream>>>(x, x16);
  }

  const size_t lh = (size_t)B_N * H_N;          // elements per (layer, state)
  hipMemsetAsync(hbuf, 0, (size_t)L_N * lh * sizeof(float), stream);
  hipMemsetAsync(hbuf16, 0, (size_t)L_N * lh * sizeof(_Float16), stream);
  hipMemsetAsync(out + (size_t)T_N * lh + (size_t)L_N * lh, 0,
                 (size_t)L_N * lh * sizeof(float), stream);

  const size_t wall_l = (size_t)H2_N * H_N;     // per-layer stride in Wx/Wh
  const size_t wsq_l  = (size_t)H_N * H_N;      // per-layer stride in Wnx/Wnh/Wtx

  for (int t = 0; t < T_N; ++t) {
    const int cur = t & 1, nxt = cur ^ 1;
    float*    h0c   = hbuf   + (size_t)cur * L_N * lh;
    float*    h0n   = hbuf   + (size_t)nxt * L_N * lh;
    _Float16* h0c16 = hbuf16 + (size_t)cur * L_N * lh;
    _Float16* h0n16 = hbuf16 + (size_t)nxt * L_N * lh;
    float*    h1c   = h0c + lh;
    float*    h1n   = h0n + lh;
    _Float16* h1c16 = h0c16 + lh;
    _Float16* h1n16 = h0n16 + lh;
    const _Float16* xt16 = x16 + (size_t)t * lh;

    // layer 0
    step_s1<<<16, 256, 0, stream>>>(xt16, h0c16, h0c, Wx, Wh, Wtx, Wnx,
                                    b_all, b_tx, b_n, zbuf, rh16, txb, nxb);
    step_s2<<<4, 256, 0, stream>>>(rh16, h0c, Wnh, zbuf, txb, nxb,
                                   h0n, h0n16, (float*)nullptr);
    // layer 1 (input = freshly-computed h0n16 / h0n)
    step_s1<<<16, 256, 0, stream>>>(h0n16, h1c16, h1c,
                                    Wx + wall_l, Wh + wall_l,
                                    Wtx + wsq_l, Wnx + wsq_l,
                                    b_all + H2_N, b_tx + H_N, b_n + H_N,
                                    zbuf, rh16, txb, nxb);
    step_s2<<<4, 256, 0, stream>>>(rh16, h1c, Wnh + wsq_l, zbuf, txb, nxb,
                                   h1n, h1n16, out + (size_t)t * lh);
  }

  // After t = T-1 (odd), final state lives in buffer 0 == hbuf base.
  hipMemcpyAsync(out + (size_t)T_N * lh, hbuf,
                 (size_t)L_N * lh * sizeof(float),
                 hipMemcpyDeviceToDevice, stream);
}
